// MultiHeadSelfAttention_55817394979302
// MI455X (gfx1250) — compile-verified, matrix operand order checked by reading, not run on hardware
//
#include <hip/hip_runtime.h>
#include <hip/hip_bf16.h>

// ---------------------------------------------------------------------------
// Multi-head self-attention, fused, bf16 WMMA pipeline for gfx1250 (MI455X).
//   1) convert inputs->bf16, weights->bf16 transposed (Wt[n][k])
//   2) QKV GEMM: async-LDS staged tiles + v_wmma_f32_16x16x32_bf16
//      (V stored transposed [B,H,hd,S])
//   3) flash-attention per (b,h): K/V tiles staged via
//      global_load_async_to_lds_b128, QK^T -> online softmax -> PV, all WMMA
// ---------------------------------------------------------------------------

#define SEQ    2048
#define DMODEL 1024
#define HEADS  16
#define HDIM   64
#define BATCH  2
#define MROWS  (BATCH * SEQ)   // 4096

typedef __attribute__((ext_vector_type(16))) __bf16 v16bf;
typedef __attribute__((ext_vector_type(8)))  float  v8f;

struct __align__(16) U4 { unsigned int x, y, z, w; };
struct __align__(32) Frag32B { U4 lo, hi; };

__device__ __forceinline__ unsigned short f32_to_bf16(float f) {
  unsigned int u = __float_as_uint(f);
  u += 0x7fffu + ((u >> 16) & 1u);   // round-to-nearest-even
  return (unsigned short)(u >> 16);
}

// Load one 16-bit A/B WMMA fragment (16 bf16 per lane): two contiguous 16-byte
// chunks at element offsets `off` and `off+16` of a 32-element K window.
__device__ __forceinline__ v16bf load_frag(const unsigned short* base, int off) {
  Frag32B p;
  p.lo = *(const U4*)(base + off);
  p.hi = *(const U4*)(base + off + 16);
  return __builtin_bit_cast(v16bf, p);
}

__device__ __forceinline__ v8f wmma_bf16(v16bf a, v16bf b, v8f c) {
  return __builtin_amdgcn_wmma_f32_16x16x32_bf16(
      false, a, false, b, (short)0, c, false, false);
}

// Async copy of one 16-byte chunk global -> LDS (ASYNCcnt-tracked).
// Low 32 bits of a generic pointer to __shared__ are the LDS byte offset.
__device__ __forceinline__ void async_b128(const unsigned short* gptr,
                                           unsigned short* lptr) {
  unsigned lds = (unsigned)(size_t)lptr;
  unsigned long long ga = (unsigned long long)(size_t)gptr;
  asm volatile("global_load_async_to_lds_b128 %0, %1, off"
               :: "v"(lds), "v"(ga) : "memory");
}

__device__ __forceinline__ void wait_async0() {
  asm volatile("s_wait_asynccnt 0x0" ::: "memory");
}

// ---------------------------------------------------------------------------
// Kernel 1a: plain f32 -> bf16 conversion
// ---------------------------------------------------------------------------
__global__ void cvt_plain_kernel(const float* __restrict__ in,
                                 unsigned short* __restrict__ out, int n) {
  int i = blockIdx.x * 256 + threadIdx.x;
  if (i < n) out[i] = f32_to_bf16(in[i]);
}

// ---------------------------------------------------------------------------
// Kernel 1b: f32 [K=1024][N=1024] -> bf16 transposed [N][K]
// ---------------------------------------------------------------------------
__global__ void cvt_transpose_kernel(const float* __restrict__ in,
                                     unsigned short* __restrict__ out) {
  int i = blockIdx.x * 256 + threadIdx.x;  // over 1024*1024
  int k = i >> 10;
  int n = i & 1023;
  out[((size_t)n << 10) + k] = f32_to_bf16(in[i]);
}

// ---------------------------------------------------------------------------
// Kernel 2: QKV projection GEMM, async-LDS staged.
//   C[4096,1024] = X16 * Wt^T. blockIdx.z selects Q / K / V.
//   Block tile 128x128, k-step 32. 8 waves; wave computes 32x64.
// ---------------------------------------------------------------------------
__global__ __launch_bounds__(256) void qkv_gemm_kernel(
    const unsigned short* __restrict__ X16,
    const unsigned short* __restrict__ WqT,
    const unsigned short* __restrict__ WkT,
    const unsigned short* __restrict__ WvT,
    unsigned short* __restrict__ Q16,
    unsigned short* __restrict__ K16,
    unsigned short* __restrict__ V16t) {
  __shared__ __align__(16) unsigned short Xlds[128 * 32];  // [row][k] 8KB
  __shared__ __align__(16) unsigned short Wlds[128 * 32];  // [col][k] 8KB

  const int tid  = threadIdx.x;
  const int lane = tid & 31;
  const int wid  = tid >> 5;
  const int half = lane >> 4;
  const int lidx = lane & 15;
  const int chunk = half * 8;

  const int which = blockIdx.z;
  const unsigned short* Wt = (which == 0) ? WqT : (which == 1) ? WkT : WvT;

  const int bm0 = blockIdx.x * 128;
  const int bn0 = blockIdx.y * 128;
  const int wm0 = (wid & 3) * 32;   // wave row offset within tile
  const int wn0 = (wid >> 2) * 64;  // wave col offset within tile

  // staging assignment: 512 chunks of 16B per tile, 2 per thread
  const int r0 = tid >> 2, p0 = (tid & 3) * 8;          // chunk A: rows 0..63
  const int r1 = (tid + 256) >> 2;                      // chunk B: rows 64..127

  v8f acc[2][4];
#pragma unroll
  for (int mt = 0; mt < 2; ++mt)
#pragma unroll
    for (int nt = 0; nt < 4; ++nt)
#pragma unroll
      for (int v = 0; v < 8; ++v) acc[mt][nt][v] = 0.0f;

  for (int k0 = 0; k0 < DMODEL; k0 += 32) {
    // ---- async stage X and Wt tiles into LDS -----------------------------
    async_b128(X16 + (size_t)(bm0 + r0) * DMODEL + k0 + p0, &Xlds[r0 * 32 + p0]);
    async_b128(X16 + (size_t)(bm0 + r1) * DMODEL + k0 + p0, &Xlds[r1 * 32 + p0]);
    async_b128(Wt  + (size_t)(bn0 + r0) * DMODEL + k0 + p0, &Wlds[r0 * 32 + p0]);
    async_b128(Wt  + (size_t)(bn0 + r1) * DMODEL + k0 + p0, &Wlds[r1 * 32 + p0]);
    wait_async0();
    __syncthreads();

    // ---- fragments from LDS + 8 WMMAs ------------------------------------
    v16bf af[2], bf[4];
#pragma unroll
    for (int mt = 0; mt < 2; ++mt)
      af[mt] = load_frag(&Xlds[(wm0 + mt * 16 + lidx) * 32], chunk);
#pragma unroll
    for (int nt = 0; nt < 4; ++nt)
      bf[nt] = load_frag(&Wlds[(wn0 + nt * 16 + lidx) * 32], chunk);
#pragma unroll
    for (int mt = 0; mt < 2; ++mt)
#pragma unroll
      for (int nt = 0; nt < 4; ++nt)
        acc[mt][nt] = wmma_bf16(af[mt], bf[nt], acc[mt][nt]);
    __syncthreads();   // protect LDS tiles before next stage
  }

  // ---- epilogue (uniform branch hoisted out of the store loops) ----------
  if (which == 2) {
#pragma unroll
    for (int mt = 0; mt < 2; ++mt)
#pragma unroll
      for (int nt = 0; nt < 4; ++nt)
#pragma unroll
        for (int v = 0; v < 8; ++v) {
          const int m = bm0 + wm0 + mt * 16 + v + half * 8;  // token
          const int n = bn0 + wn0 + nt * 16 + lidx;          // embed col
          const int b = m >> 11, s = m & 2047;
          const int h = n >> 6,  d = n & 63;
          V16t[(((size_t)(b * HEADS + h)) * HDIM + d) * SEQ + s] =
              f32_to_bf16(acc[mt][nt][v]);
        }
  } else {
    unsigned short* dst = (which == 0) ? Q16 : K16;
#pragma unroll
    for (int mt = 0; mt < 2; ++mt)
#pragma unroll
      for (int nt = 0; nt < 4; ++nt)
#pragma unroll
        for (int v = 0; v < 8; ++v) {
          const int m = bm0 + wm0 + mt * 16 + v + half * 8;
          const int n = bn0 + wn0 + nt * 16 + lidx;
          dst[(size_t)m * DMODEL + n] = f32_to_bf16(acc[mt][nt][v]);
        }
  }
}

// ---------------------------------------------------------------------------
// Kernel 3: flash attention. grid = (SEQ/128, BATCH*HEADS), 256 threads.
// K/V tiles for each 32-key block staged cooperatively via async-LDS.
// ---------------------------------------------------------------------------
__global__ __launch_bounds__(256) void attn_kernel(
    const unsigned short* __restrict__ Q16,
    const unsigned short* __restrict__ K16,
    const unsigned short* __restrict__ V16t,
    const float* __restrict__ mask,
    float* __restrict__ out) {
  __shared__ __align__(16) unsigned short Klds[32 * 64];    // [key][d]   4KB
  __shared__ __align__(16) unsigned short Vlds[64 * 32];    // [d][key]   4KB
  __shared__ __align__(16) unsigned short Plds[8][16 * 32]; // per-wave P 8KB

  const int tid  = threadIdx.x;
  const int lane = tid & 31;
  const int wid  = tid >> 5;
  const int half = lane >> 4;
  const int lidx = lane & 15;
  const int chunk = half * 8;

  const int bh = blockIdx.y;        // 0..31
  const int b  = bh >> 4;
  const int h  = bh & 15;
  const int q0 = blockIdx.x * 128 + wid * 16;

  const float* maskp = mask + (size_t)b * SEQ;

  // staging assignment for K tile (32 keys x 64 d = 256 chunks of 16B)
  const int kkey = tid >> 3, kpart = (tid & 7) * 8;
  // staging assignment for V tile (64 d x 32 keys = 256 chunks of 16B)
  const int vd = tid >> 2, vpart = (tid & 3) * 8;
  const unsigned short* Kbase =
      K16 + (size_t)(b * SEQ + kkey) * DMODEL + h * HDIM + kpart;
  const unsigned short* Vbase = V16t + (size_t)(bh * HDIM + vd) * SEQ + vpart;

  // Preload Q A-fragments: 16x64 as two 16x32 frags (per-wave, direct global).
  const unsigned short* Qrow =
      Q16 + (size_t)(b * SEQ + q0 + lidx) * DMODEL + h * HDIM;
  const v16bf qf0 = load_frag(Qrow, chunk);
  const v16bf qf1 = load_frag(Qrow + 32, chunk);

  float mrow[8], lrow[8];
  v8f acc[4];
#pragma unroll
  for (int v = 0; v < 8; ++v) { mrow[v] = -1e30f; lrow[v] = 0.0f; }
#pragma unroll
  for (int n = 0; n < 4; ++n)
#pragma unroll
    for (int v = 0; v < 8; ++v) acc[n][v] = 0.0f;

  for (int kb = 0; kb < SEQ; kb += 32) {
    // ---- async stage K (32x64) and V^T (64x32) tiles ---------------------
    async_b128(Kbase + (size_t)kb * DMODEL, &Klds[kkey * 64 + kpart]);
    async_b128(Vbase + kb, &Vlds[vd * 32 + vpart]);
    __builtin_prefetch(Kbase + (size_t)(kb + 32) * DMODEL, 0, 1);
    wait_async0();
    __syncthreads();

    // ---- scores: Q(16x64) x K^T(64x32) as two 16-key N tiles -------------
    v8f sA, sB;
#pragma unroll
    for (int v = 0; v < 8; ++v) { sA[v] = 0.0f; sB[v] = 0.0f; }
    sA = wmma_bf16(qf0, load_frag(&Klds[lidx * 64], chunk), sA);
    sA = wmma_bf16(qf1, load_frag(&Klds[lidx * 64], 32 + chunk), sA);
    sB = wmma_bf16(qf0, load_frag(&Klds[(16 + lidx) * 64], chunk), sB);
    sB = wmma_bf16(qf1, load_frag(&Klds[(16 + lidx) * 64], 32 + chunk), sB);

    const float addA = maskp[kb + lidx] * -1e9f;
    const float addB = maskp[kb + 16 + lidx] * -1e9f;

    // ---- online softmax over the 32 new keys -----------------------------
#pragma unroll
    for (int v = 0; v < 8; ++v) {
      float a = sA[v] * 0.125f + addA;   // 1/sqrt(64)
      float c = sB[v] * 0.125f + addB;
      float x = fmaxf(a, c);
      x = fmaxf(x, __shfl_xor(x, 1, 32));
      x = fmaxf(x, __shfl_xor(x, 2, 32));
      x = fmaxf(x, __shfl_xor(x, 4, 32));
      x = fmaxf(x, __shfl_xor(x, 8, 32));
      const float mnew = fmaxf(mrow[v], x);
      const float corr = __expf(mrow[v] - mnew);
      mrow[v] = mnew;
      const float pA = __expf(a - mnew);
      const float pB = __expf(c - mnew);
      float rs = pA + pB;
      rs += __shfl_xor(rs, 1, 32);
      rs += __shfl_xor(rs, 2, 32);
      rs += __shfl_xor(rs, 4, 32);
      rs += __shfl_xor(rs, 8, 32);
      lrow[v] = lrow[v] * corr + rs;
#pragma unroll
      for (int n = 0; n < 4; ++n) acc[n][v] *= corr;
      // park P in LDS (bf16) to re-load in A-fragment layout
      const int row = v + half * 8;
      Plds[wid][row * 32 + lidx]      = f32_to_bf16(pA);
      Plds[wid][row * 32 + 16 + lidx] = f32_to_bf16(pB);
    }

    // ---- PV: P(16x32) x V(32x64) -----------------------------------------
    const v16bf pf = load_frag(&Plds[wid][lidx * 32], chunk);
#pragma unroll
    for (int n = 0; n < 4; ++n)
      acc[n] = wmma_bf16(pf, load_frag(&Vlds[(n * 16 + lidx) * 32], chunk),
                         acc[n]);
    __syncthreads();   // protect K/V tiles before next stage
  }

  // ---- epilogue: out[b, q, h*64 + d] = acc / l ---------------------------
#pragma unroll
  for (int v = 0; v < 8; ++v) {
    const float inv = 1.0f / lrow[v];
    const int q = q0 + v + half * 8;
    float* orow = out + (size_t)(b * SEQ + q) * DMODEL + h * HDIM;
#pragma unroll
    for (int n = 0; n < 4; ++n) orow[n * 16 + lidx] = acc[n][v] * inv;
  }
}

// ---------------------------------------------------------------------------
extern "C" void kernel_launch(void* const* d_in, const int* in_sizes, int n_in,
                              void* d_out, int out_size, void* d_ws,
                              size_t ws_size, hipStream_t stream) {
  const float* x    = (const float*)d_in[0];  // [2,2048,1024]
  const float* mask = (const float*)d_in[1];  // [2,2048]
  const float* wq   = (const float*)d_in[2];  // [1024,1024]
  const float* wk   = (const float*)d_in[3];
  const float* wv   = (const float*)d_in[4];
  float* out = (float*)d_out;

  // Workspace layout (bf16 = unsigned short), total ~38 MB.
  unsigned short* X16 = (unsigned short*)d_ws;
  unsigned short* WqT = X16 + (size_t)MROWS * DMODEL;
  unsigned short* WkT = WqT + (size_t)DMODEL * DMODEL;
  unsigned short* WvT = WkT + (size_t)DMODEL * DMODEL;
  unsigned short* Q16 = WvT + (size_t)DMODEL * DMODEL;
  unsigned short* K16 = Q16 + (size_t)MROWS * DMODEL;
  unsigned short* V16t = K16 + (size_t)MROWS * DMODEL;

  {
    const int n = MROWS * DMODEL;
    cvt_plain_kernel<<<(n + 255) / 256, 256, 0, stream>>>(x, X16, n);
    const int nw = DMODEL * DMODEL;
    cvt_transpose_kernel<<<nw / 256, 256, 0, stream>>>(wq, WqT);
    cvt_transpose_kernel<<<nw / 256, 256, 0, stream>>>(wk, WkT);
    cvt_transpose_kernel<<<nw / 256, 256, 0, stream>>>(wv, WvT);
  }

  qkv_gemm_kernel<<<dim3(MROWS / 128, DMODEL / 128, 3), 256, 0, stream>>>(
      X16, WqT, WkT, WvT, Q16, K16, V16t);

  attn_kernel<<<dim3(SEQ / 128, BATCH * HEADS), 256, 0, stream>>>(
      Q16, K16, V16t, mask, out);
}
